// MultiRateLSTM_18176301596876
// MI455X (gfx1250) — matrix operand order repaired and checked by reading, not compiled, not run on hardware
//
#include <hip/hip_runtime.h>
#include <hip/hip_bf16.h>
#include <math.h>

// ---------------------------------------------------------------------------
// MultiRateLSTM on gfx1250 (MI455X)
//   T=8192, B=16, H=256, gates=1024.
//   Combined K = 256 (x image) + 256 (h) + 32 (bias chunk, A==1 at k=512) = 544
//   Persistent single-WGP scan, bf16 WMMA 16x16x32, double-buffered A image,
//   one barrier per step, bias folded into the GEMM (first WMMA uses C=0),
//   fast TRANS-unit sigmoid/tanh, and explicit 2-deep register pipelining of
//   the weight (B) stream that runs across the step barrier.
// ---------------------------------------------------------------------------

typedef __attribute__((ext_vector_type(16))) __bf16 v16bf;
typedef __attribute__((ext_vector_type(8)))  __bf16 v8bf;
typedef __attribute__((ext_vector_type(8)))  float  v8f;

#define T_STEPS   8192
#define BATCH     16
#define HID       256
#define NT        64                 // 1024 / 16 n-tiles
#define KC        17                 // 544 / 32 k-chunks (last = bias chunk)
#define TILE_EL   512                // 32 lanes * 16 bf16 per (nt,kc) tile
#define CELL_EL   (NT*KC*TILE_EL)    // 557056 bf16 per cell
#define ABUF_EL   (KC*TILE_EL)       // 8704 bf16 per A buffer

// workspace layout (bytes)
#define WPACK_OFF   0
#define WPACK_BYTES (3*CELL_EL*2)
#define X0B_OFF     (WPACK_OFF + WPACK_BYTES)
#define X0B_EL      (8192*16*64)
#define X1B_OFF     (X0B_OFF + X0B_EL*2)
#define X1B_EL      (4096*16*64)
#define X2B_OFF     (X1B_OFF + X1B_EL*2)
#define X2B_EL      (2048*16*128)

// ---------------------------------------------------------------------------
// Pack W_comb (zero-padded to K=544, bias in row k=512) into the exact 32x16
// bf16 B-operand layout: element e of lane L in tile (nt,kc) is
//   n = nt*16 + (L&15),  k = kc*32 + (e>>3)*16 + (L>>4)*8 + (e&7)
// stored as wpack[cell][(nt*KC+kc)*512 + L*16 + e].
// Combined-K map: [0:64)=x0, [64:128)=x1, [128:256)=x2, [256:512)=h, 512=bias.
// ---------------------------------------------------------------------------
__global__ void mrl_pack_weights(const float* __restrict__ Wih_a, const float* __restrict__ Whh_a,
                                 const float* __restrict__ bi_a,  const float* __restrict__ bh_a,
                                 const float* __restrict__ Wih_b, const float* __restrict__ Whh_b,
                                 const float* __restrict__ bi_b,  const float* __restrict__ bh_b,
                                 const float* __restrict__ Wih_c, const float* __restrict__ Whh_c,
                                 const float* __restrict__ bi_c,  const float* __restrict__ bh_c,
                                 __bf16* __restrict__ wpack) {
  int idx = blockIdx.x * blockDim.x + threadIdx.x;
  if (idx >= 3 * CELL_EL) return;
  int cell   = idx / CELL_EL;
  int rem    = idx % CELL_EL;
  int tile   = rem / TILE_EL;
  int within = rem % TILE_EL;
  int lane = within >> 4, e = within & 15;
  int nt = tile / KC, kc = tile % KC;
  int n = nt * 16 + (lane & 15);
  int k = kc * 32 + (e >> 3) * 16 + (lane >> 4) * 8 + (e & 7);
  float v = 0.0f;
  if (k == 512) {                       // bias row (A carries 1.0 here)
    const float* bi = (cell == 0) ? bi_a : (cell == 1) ? bi_b : bi_c;
    const float* bh = (cell == 0) ? bh_a : (cell == 1) ? bh_b : bh_c;
    v = bi[n] + bh[n];
  } else if (k > 512) {
    v = 0.0f;
  } else if (k >= 256) {                // recurrent part (all cells, H=256)
    const float* Whh = (cell == 0) ? Whh_a : (cell == 1) ? Whh_b : Whh_c;
    v = Whh[n * 256 + (k - 256)];
  } else if (cell == 0) {               // cell a: full 256-wide input
    v = Wih_a[n * 256 + k];
  } else if (cell == 1) {               // cell b: x0|x1 (128), zero elsewhere
    if (k < 128) v = Wih_b[n * 128 + k];
  } else {                              // cell c: x0 only (64), zero elsewhere
    if (k < 64)  v = Wih_c[n * 64 + k];
  }
  wpack[idx] = (__bf16)v;
}

// f32 -> bf16 convert for the x streams
__global__ void mrl_cvt_bf16(const float* __restrict__ in, __bf16* __restrict__ out, int n) {
  int i = blockIdx.x * blockDim.x + threadIdx.x;
  if (i < n) out[i] = (__bf16)in[i];
}

// ---------------------------------------------------------------------------
// Fast TRANS-unit activations (v_exp2 + v_rcp co-execute with the XDL pipe).
// ---------------------------------------------------------------------------
#define MRL_NLOG2E  (-1.4426950408889634f)
__device__ __forceinline__ float mrl_sigmoid(float x) {
  return __builtin_amdgcn_rcpf(1.0f + __builtin_amdgcn_exp2f(x * MRL_NLOG2E));
}
__device__ __forceinline__ float mrl_tanh(float x) {
  float r = __builtin_amdgcn_rcpf(1.0f + __builtin_amdgcn_exp2f(x * (2.0f * MRL_NLOG2E)));
  return __builtin_fmaf(2.0f, r, -1.0f);
}

// ---------------------------------------------------------------------------
// Persistent scan kernel: 1 block, 512 threads = 16 waves. Wave w owns hidden
// columns [16w, 16w+16); its 4 n-tiles (i/f/g/o) are nt = gi*16 + w.
// A operand ([16 x 544] bf16, double-buffered) lives in LDS in the A-matrix
// VGPR image:
//   Abuf[kc*512 + L*16 + e] <=> A[m=L&15][k = kc*32 + (e>>3)*16 + (L>>4)*8 + (e&7)]
// B (weight) stream: explicit 2-deep register pipeline; the last two stages of
// each step issue the NEXT step's chunk-0/1 loads, which remain in flight
// across the barrier and the elementwise tail.
// ---------------------------------------------------------------------------
__device__ __forceinline__ v16bf mrl_ldb(const __bf16* Wlane, int tileIdx) {
  return *(const v16bf*)(Wlane + ((size_t)tileIdx << 9));
}

__launch_bounds__(512, 1)
__global__ void mrl_scan(const __bf16* __restrict__ wpack,
                         const __bf16* __restrict__ x0b,
                         const __bf16* __restrict__ x1b,
                         const __bf16* __restrict__ x2b,
                         float* __restrict__ out) {
  __shared__ __align__(16) __bf16 Alds[2 * ABUF_EL];   // 34 KB, double-buffered

  const int tid    = threadIdx.x;
  const int lane   = tid & 31;
  const int w      = tid >> 5;            // wave id 0..15
  const int n_lane = lane & 15;
  const int mbase  = (lane >> 4) * 8;     // C/D layout: m = mbase + vgpr

  // ---- init: zero both buffers, then plant the 1.0 bias column (k=512) ----
  for (int i = tid; i < 2 * ABUF_EL; i += 512) Alds[i] = (__bf16)0.0f;
  __syncthreads();
  if (tid < 32) {                          // k=512 -> kc=16, e=0, L=m
    int m = tid & 15, b = tid >> 4;
    Alds[b * ABUF_EL + 16 * 512 + m * 16] = (__bf16)1.0f;
  }

  // ---- per-thread x-staging constants: one aligned 8-elem run per thread ----
  const int i0 = tid * 8;
  const int s  = (i0 < 1024) ? 0 : ((i0 < 2048) ? 1 : 2);
  const int j  = i0 - ((s == 0) ? 0 : (s == 1) ? 1024 : 2048);
  const int m_x = (s == 2) ? (j >> 7) : (j >> 6);
  const int c0  = (s == 2) ? (j & 127) : (j & 63);
  const int k0  = (s == 0) ? c0 : ((s == 1) ? (64 + c0) : (128 + c0));
  const int kcx = k0 >> 5, r0 = k0 & 31;
  const int e0  = ((r0 >> 4) & 1) * 8;                 // r0&7 == 0 (8-aligned)
  const int Lx  = ((r0 >> 3) & 1) * 16 + m_x;
  const int xdst = kcx * 512 + Lx * 16 + e0;           // 16B-aligned
  const __bf16* sbase  = ((s == 0) ? x0b : (s == 1) ? x1b : x2b) + j;
  const int     xshift = s;                            // t>>0 / t>>1 / t>>2
  const int     xslice = (s == 2) ? 2048 : 1024;

  // ---- per-wave h-staging constants (k = 256 + 16w + n_lane) ----
  const int kh  = 256 + w * 16 + n_lane;
  const int kch = kh >> 5, rh = kh & 31;
  const int eh  = ((rh >> 4) & 1) * 8 + (rh & 7);
  const int Lbh = ((rh >> 3) & 1) * 16;

  float hloc[8];
  v8f   cst;
#pragma unroll
  for (int v = 0; v < 8; ++v) { hloc[v] = 0.0f; cst[v] = 0.0f; }

  // ---- prime the B pipeline with step 0 (sel=0) chunks 0 and 1 ----
  const __bf16* W0 = wpack + lane * 16;
  v16bf b0[4], b1[4], b2[4];
#pragma unroll
  for (int g = 0; g < 4; ++g) {
    b0[g] = mrl_ldb(W0, (g * 16 + w) * KC + 0);
    b1[g] = mrl_ldb(W0, (g * 16 + w) * KC + 1);
  }

  for (int t = 0; t < T_STEPS; ++t) {
    __bf16* Ab = &Alds[(t & 1) * ABUF_EL];

    // ---- stage: one 16B x load+store, 8 scalar h stores ----
    *(v8bf*)(&Ab[xdst]) = *(const v8bf*)(sbase + (size_t)(t >> xshift) * xslice);
#pragma unroll
    for (int v = 0; v < 8; ++v)
      Ab[kch * 512 + (Lbh + mbase + v) * 16 + eh] = (__bf16)hloc[v];
    __syncthreads();

    // ---- cell for this step / next step ----
    const int sel  = ((t & 3) == 0) ? 0 : (((t & 1) == 0) ? 1 : 2);
    const int tn   = t + 1;
    const int seln = ((tn & 3) == 0) ? 0 : (((tn & 1) == 0) ? 1 : 2);
    const __bf16* Wsel = wpack + (size_t)sel  * CELL_EL + lane * 16;
    const __bf16* Wnx  = wpack + (size_t)seln * CELL_EL + lane * 16;

    // ---- gates = A[16,544] x W[544, 4x16]; 68 WMMAs, 2-deep B pipeline ----
    v8f acc0, acc1, acc2, acc3;
    v16bf a0 = *(const v16bf*)(&Ab[lane * 16]);
    v16bf a1;
#pragma unroll
    for (int kc = 0; kc < KC; ++kc) {
      // issue chunk kc+2 (or next step's chunk 0/1) loads two stages ahead
      {
        const bool in_step = (kc + 2 < KC);
        const __bf16* Ws = in_step ? Wsel : Wnx;
        const int kcn    = in_step ? (kc + 2) : (kc + 2 - KC);
#pragma unroll
        for (int g = 0; g < 4; ++g)
          b2[g] = mrl_ldb(Ws, (g * 16 + w) * KC + kcn);
      }
      if (kc + 1 < KC)
        a1 = *(const v16bf*)(&Ab[(kc + 1) * 512 + lane * 16]);

      if (kc == 0) {
        v8f z = {};                                       // SRC2 = inline 0
        acc0 = __builtin_amdgcn_wmma_f32_16x16x32_bf16(false, a0, false, b0[0], (short)0, z, false, false);
        acc1 = __builtin_amdgcn_wmma_f32_16x16x32_bf16(false, a0, false, b0[1], (short)0, z, false, false);
        acc2 = __builtin_amdgcn_wmma_f32_16x16x32_bf16(false, a0, false, b0[2], (short)0, z, false, false);
        acc3 = __builtin_amdgcn_wmma_f32_16x16x32_bf16(false, a0, false, b0[3], (short)0, z, false, false);
      } else {
        acc0 = __builtin_amdgcn_wmma_f32_16x16x32_bf16(false, a0, false, b0[0], (short)0, acc0, false, false);
        acc1 = __builtin_amdgcn_wmma_f32_16x16x32_bf16(false, a0, false, b0[1], (short)0, acc1, false, false);
        acc2 = __builtin_amdgcn_wmma_f32_16x16x32_bf16(false, a0, false, b0[2], (short)0, acc2, false, false);
        acc3 = __builtin_amdgcn_wmma_f32_16x16x32_bf16(false, a0, false, b0[3], (short)0, acc3, false, false);
      }
      // rotate pipeline registers (SSA after full unroll)
      a0 = a1;
#pragma unroll
      for (int g = 0; g < 4; ++g) { b0[g] = b1[g]; b1[g] = b2[g]; }
    }
    // b0/b1 now hold NEXT step's chunks 0 and 1 (loads already in flight).

    // ---- elementwise LSTM update + output write (c stays in VGPRs) ----
    float* orow = out + (size_t)t * (BATCH * HID) + w * 16 + n_lane;
#pragma unroll
    for (int v = 0; v < 8; ++v) {
      float ig = mrl_sigmoid(acc0[v]);
      float fg = mrl_sigmoid(acc1[v]);
      float gg = mrl_tanh(acc2[v]);
      float og = mrl_sigmoid(acc3[v]);
      float cn = __builtin_fmaf(fg, cst[v], ig * gg);
      cst[v]  = cn;
      float hn = og * mrl_tanh(cn);
      hloc[v] = hn;
      orow[(size_t)(mbase + v) * HID] = hn;
    }

    // ---- prefetch the rest of next step's weight tiles into the caches ----
    {
      const __bf16* Wn = wpack + (size_t)seln * CELL_EL;
#pragma unroll
      for (int gi = 0; gi < 4; ++gi)
        __builtin_prefetch(Wn + (size_t)(((gi * 16 + w) * KC) << 9) + lane * 272, 0, 0);
    }
    // no second barrier: next iteration stages into the other buffer
  }
}

// ---------------------------------------------------------------------------
extern "C" void kernel_launch(void* const* d_in, const int* in_sizes, int n_in,
                              void* d_out, int out_size, void* d_ws, size_t ws_size,
                              hipStream_t stream) {
  const float* x0    = (const float*)d_in[0];
  const float* x1    = (const float*)d_in[1];
  const float* x2    = (const float*)d_in[2];
  const float* Wih_a = (const float*)d_in[3];
  const float* Whh_a = (const float*)d_in[4];
  const float* bih_a = (const float*)d_in[5];
  const float* bhh_a = (const float*)d_in[6];
  const float* Wih_b = (const float*)d_in[7];
  const float* Whh_b = (const float*)d_in[8];
  const float* bih_b = (const float*)d_in[9];
  const float* bhh_b = (const float*)d_in[10];
  const float* Wih_c = (const float*)d_in[11];
  const float* Whh_c = (const float*)d_in[12];
  const float* bih_c = (const float*)d_in[13];
  const float* bhh_c = (const float*)d_in[14];

  char* ws = (char*)d_ws;
  __bf16* wpack = (__bf16*)(ws + WPACK_OFF);
  __bf16* x0b   = (__bf16*)(ws + X0B_OFF);
  __bf16* x1b   = (__bf16*)(ws + X1B_OFF);
  __bf16* x2b   = (__bf16*)(ws + X2B_OFF);

  mrl_pack_weights<<<(3 * CELL_EL + 255) / 256, 256, 0, stream>>>(
      Wih_a, Whh_a, bih_a, bhh_a,
      Wih_b, Whh_b, bih_b, bhh_b,
      Wih_c, Whh_c, bih_c, bhh_c, wpack);
  mrl_cvt_bf16<<<(X0B_EL + 255) / 256, 256, 0, stream>>>(x0, x0b, X0B_EL);
  mrl_cvt_bf16<<<(X1B_EL + 255) / 256, 256, 0, stream>>>(x1, x1b, X1B_EL);
  mrl_cvt_bf16<<<(X2B_EL + 255) / 256, 256, 0, stream>>>(x2, x2b, X2B_EL);

  mrl_scan<<<1, 512, 0, stream>>>(wpack, x0b, x1b, x2b, (float*)d_out);
}
